// ContraNorm_84232898609350
// MI455X (gfx1250) — matrix operand order repaired and checked by reading, not compiled, not run on hardware
//
#include <hip/hip_runtime.h>

// ---------------------------------------------------------------------------
// ContraNorm for MI455X (gfx1250): flash-attention formulation, f16 WMMA.
// out = 1.2*x - 0.2 * softmax( (x x^T) * rn_i * rn_j ) * x,  rn = 1/max(||x||,eps)
// Q rows pre-scaled by rn_i (registers), K rows pre-scaled by rn_j (LDS),
// V raw (transposed LDS copy) -> S from WMMA is already the scaled similarity.
// ---------------------------------------------------------------------------

typedef __attribute__((ext_vector_type(16))) _Float16 v16h;
typedef __attribute__((ext_vector_type(8)))  _Float16 v8h;
typedef __attribute__((ext_vector_type(4)))  _Float16 v4h;
typedef __attribute__((ext_vector_type(8)))  float    v8f;

#define DIM       512
#define NSEQ      2048
#define NBATCH    8
#define BM        32            // query rows per workgroup (2 m-tiles)
#define BJ        32            // key rows per iteration (2 j-tiles)
#define NWAVE     8             // waves per workgroup; each owns 64 cols of D
#define KV_STRIDE 520           // halfs per row (512 + 8 pad) -> 1040B, 16B aligned
#define VT_STRIDE 40            // halfs per transposed row (32 + 8 pad) -> 80B
#define S_STRIDE  36            // floats per S row (32 + 4 pad), 144B
#define S_SLAB    (BM * S_STRIDE)
#define SCALE     0.2f

union V16 {
    v16h v;
    v8h  h[2];
    _Float16 e[16];
};

// ---------------- Kernel 1: inverse row norms --------------------------------
__global__ __launch_bounds__(256) void rownorm_kernel(const float* __restrict__ x,
                                                      float* __restrict__ rn) {
    const int row  = blockIdx.x * 8 + (threadIdx.x >> 5);
    const int lane = threadIdx.x & 31;
    const float* p = x + (size_t)row * DIM + lane * 16;
    float s = 0.f;
#pragma unroll
    for (int i = 0; i < 4; ++i) {
        float4 v = *(const float4*)(p + i * 4);
        s += v.x * v.x + v.y * v.y + v.z * v.z + v.w * v.w;
    }
#pragma unroll
    for (int off = 16; off; off >>= 1) s += __shfl_xor(s, off, 32);
    if (lane == 0) rn[row] = 1.0f / fmaxf(sqrtf(s), 1e-12f);
}

// ---------------- Kernel 2: fused attention ---------------------------------
__global__ __launch_bounds__(256) void contranorm_kernel(const float* __restrict__ x,
                                                         const float* __restrict__ rn,
                                                         float* __restrict__ out) {
    __shared__ __align__(16) _Float16 sKV[BJ * KV_STRIDE];   // rn_j-scaled keys, row major
    __shared__ __align__(16) _Float16 sVT[DIM * VT_STRIDE];  // raw values, transposed
    __shared__ __align__(16) float    sSp[NWAVE * S_SLAB];   // per-wave partial S slabs

    const int b     = blockIdx.x / (NSEQ / BM);
    const int i0    = (blockIdx.x % (NSEQ / BM)) * BM;
    const int tid   = threadIdx.x;
    const int wave  = tid >> 5;
    const int lane  = tid & 31;
    const int l16   = lane & 15;
    const int lhalf = lane >> 4;          // 0 or 1
    const int kbase = lhalf * 8;          // A-operand half base

    const float* xb = x + (size_t)b * NSEQ * DIM;

    // ---- Q A-operands in registers: wave-private K slice [wave*64, wave*64+64)
    // A layout (16-bit, 16x32): lane l -> row l%16; halves K = kbase+{0..7,16..23}
    V16 Aq[2][2];
#pragma unroll
    for (int mt = 0; mt < 2; ++mt) {
        const int row = i0 + mt * 16 + l16;
        const float rni = rn[b * NSEQ + row];
        const float* xr = xb + (size_t)row * DIM;
#pragma unroll
        for (int ks = 0; ks < 2; ++ks) {
            const int k0 = wave * 64 + ks * 32 + kbase;
#pragma unroll
            for (int g = 0; g < 2; ++g) {
                float4 f0 = *(const float4*)(xr + k0 + g * 16);
                float4 f1 = *(const float4*)(xr + k0 + g * 16 + 4);
                const int e = g * 8;
                Aq[mt][ks].e[e + 0] = (_Float16)(f0.x * rni);
                Aq[mt][ks].e[e + 1] = (_Float16)(f0.y * rni);
                Aq[mt][ks].e[e + 2] = (_Float16)(f0.z * rni);
                Aq[mt][ks].e[e + 3] = (_Float16)(f0.w * rni);
                Aq[mt][ks].e[e + 4] = (_Float16)(f1.x * rni);
                Aq[mt][ks].e[e + 5] = (_Float16)(f1.y * rni);
                Aq[mt][ks].e[e + 6] = (_Float16)(f1.z * rni);
                Aq[mt][ks].e[e + 7] = (_Float16)(f1.w * rni);
            }
        }
    }

    v8f  O[2][4] = {};                    // 32 rows x 64 cols accumulator
    float mrow[2] = {-1e30f, -1e30f};     // running max (row = mt*16 + l16)
    float lrow[2] = {0.f, 0.f};           // running denom

    for (int j0 = 0; j0 < NSEQ; j0 += BJ) {
        __syncthreads();   // previous iteration's LDS reads complete

        // ---- cooperative load: 4 micro-tiles of 4x4 per thread
        //      K row-major (scaled by rn_j) + V transposed (raw), all b64 stores
        if (j0 + BJ < NSEQ) {
            __builtin_prefetch(xb + (size_t)(j0 + BJ + (tid >> 3)) * DIM + (tid & 7) * 64, 0, 1);
        }
#pragma unroll
        for (int i = 0; i < 4; ++i) {
            const int T   = tid + 256 * i;
            const int jr0 = (T >> 7) * 4;        // key-row group base (0..28)
            const int c0  = (T & 127) * 4;       // column base (0..508)
            float g[4][4];
#pragma unroll
            for (int rr = 0; rr < 4; ++rr) {
                const float4 f = *(const float4*)(xb + (size_t)(j0 + jr0 + rr) * DIM + c0);
                g[rr][0] = f.x; g[rr][1] = f.y; g[rr][2] = f.z; g[rr][3] = f.w;
            }
            // K: scale by rn_j, store row-major
#pragma unroll
            for (int rr = 0; rr < 4; ++rr) {
                const float rj = rn[b * NSEQ + j0 + jr0 + rr];   // broadcast within wave
                v4h hk = { (_Float16)(g[rr][0] * rj), (_Float16)(g[rr][1] * rj),
                           (_Float16)(g[rr][2] * rj), (_Float16)(g[rr][3] * rj) };
                *(v4h*)(sKV + (jr0 + rr) * KV_STRIDE + c0) = hk;
            }
            // V: raw, transposed
#pragma unroll
            for (int cc = 0; cc < 4; ++cc) {
                v4h hv = { (_Float16)g[0][cc], (_Float16)g[1][cc],
                           (_Float16)g[2][cc], (_Float16)g[3][cc] };
                *(v4h*)(sVT + (c0 + cc) * VT_STRIDE + jr0) = hv;
            }
        }
        __syncthreads();

        // ---- partial S = Qhat . Khat^T over this wave's K slice (f32 accum)
        // B layout (16-bit, 32x16): lane l -> col l%16; halves K = 8*lhalf + {0..7,16..23}
        v8f Sacc[2][2] = {};
#pragma unroll
        for (int ks = 0; ks < 2; ++ks)
#pragma unroll
            for (int jt = 0; jt < 2; ++jt) {
                V16 Bk;
                const _Float16* kp = sKV + (jt * 16 + l16) * KV_STRIDE
                                   + wave * 64 + ks * 32 + lhalf * 8;
                Bk.h[0] = *(const v8h*)(kp);
                Bk.h[1] = *(const v8h*)(kp + 16);
#pragma unroll
                for (int mt = 0; mt < 2; ++mt)
                    Sacc[mt][jt] = __builtin_amdgcn_wmma_f32_16x16x32_f16(
                            false, Aq[mt][ks].v, false, Bk.v, (short)0,
                            Sacc[mt][jt], false, false);
            }
        {
            float* slab = sSp + wave * S_SLAB;
#pragma unroll
            for (int mt = 0; mt < 2; ++mt)
#pragma unroll
                for (int jt = 0; jt < 2; ++jt)
#pragma unroll
                    for (int v = 0; v < 8; ++v) {
                        const int r = mt * 16 + v + 8 * lhalf;  // C layout row
                        slab[r * S_STRIDE + jt * 16 + l16] = Sacc[mt][jt][v];
                    }
        }
        __syncthreads();

        // ---- deterministic cross-wave reduction: one float4 per thread
        {
            const int r  = tid >> 3;
            const int c4 = (tid & 7) << 2;
            float4 acc = *(const float4*)(sSp + r * S_STRIDE + c4);
#pragma unroll
            for (int w = 1; w < NWAVE; ++w) {
                const float4 p = *(const float4*)(sSp + w * S_SLAB + r * S_STRIDE + c4);
                acc.x += p.x; acc.y += p.y; acc.z += p.z; acc.w += p.w;
            }
            *(float4*)(sSp + r * S_STRIDE + c4) = acc;
        }
        __syncthreads();

        // ---- online softmax, P built directly in WMMA A-operand layout
        V16  P[2];
        float alpha[2];
#pragma unroll
        for (int mt = 0; mt < 2; ++mt) {
            const float* srow = sSp + (mt * 16 + l16) * S_STRIDE;
            const float4 s0 = *(const float4*)(srow + kbase);
            const float4 s1 = *(const float4*)(srow + kbase + 4);
            const float4 s2 = *(const float4*)(srow + kbase + 16);
            const float4 s3 = *(const float4*)(srow + kbase + 20);
            float sv[16] = { s0.x, s0.y, s0.z, s0.w,  s1.x, s1.y, s1.z, s1.w,
                             s2.x, s2.y, s2.z, s2.w,  s3.x, s3.y, s3.z, s3.w };
            float mx = -1e30f;
#pragma unroll
            for (int e = 0; e < 16; ++e) mx = fmaxf(mx, sv[e]);
            mx = fmaxf(mx, __shfl_xor(mx, 16, 32));          // combine half-rows
            const float mnew = fmaxf(mrow[mt], mx);
            const float a = __expf(mrow[mt] - mnew);
            float psum = 0.f;
#pragma unroll
            for (int e = 0; e < 16; ++e) {
                const float p = __expf(sv[e] - mnew);
                P[mt].e[e] = (_Float16)p;
                psum += p;
            }
            psum += __shfl_xor(psum, 16, 32);
            lrow[mt] = lrow[mt] * a + psum;
            mrow[mt] = mnew;
            alpha[mt] = a;
        }

        // ---- rescale O by alpha (alpha lives at lane row%16 -> per-VGPR broadcast)
#pragma unroll
        for (int mt = 0; mt < 2; ++mt)
#pragma unroll
            for (int v = 0; v < 8; ++v) {
                const float av = __shfl(alpha[mt], v + 8 * lhalf, 32);
#pragma unroll
                for (int nt = 0; nt < 4; ++nt) O[mt][nt][v] *= av;
            }

        // ---- O += P . V  (B operand from transposed V rows, halfs 8*lhalf+{0..7,16..23})
#pragma unroll
        for (int nt = 0; nt < 4; ++nt) {
            V16 Bv;
            const _Float16* vp = sVT + (wave * 64 + nt * 16 + l16) * VT_STRIDE
                               + lhalf * 8;
            Bv.h[0] = *(const v8h*)(vp);
            Bv.h[1] = *(const v8h*)(vp + 16);
#pragma unroll
            for (int mt = 0; mt < 2; ++mt)
                O[mt][nt] = __builtin_amdgcn_wmma_f32_16x16x32_f16(
                        false, P[mt].v, false, Bv.v, (short)0, O[mt][nt], false, false);
        }
    }

    // ---- epilogue: out = 1.2*x - 0.2 * O/l
#pragma unroll
    for (int mt = 0; mt < 2; ++mt) {
        const float linv_mine = 1.0f / lrow[mt];
#pragma unroll
        for (int v = 0; v < 8; ++v) {
            const float linv = __shfl(linv_mine, v + 8 * lhalf, 32);
            const int row = i0 + mt * 16 + v + 8 * lhalf;
#pragma unroll
            for (int nt = 0; nt < 4; ++nt) {
                const int col = wave * 64 + nt * 16 + l16;
                const size_t idx = ((size_t)(b * NSEQ + row)) * DIM + col;
                const float xv = x[idx];
                out[idx] = (1.0f + SCALE) * xv - SCALE * (O[mt][nt][v] * linv);
            }
        }
    }
}

// ---------------------------------------------------------------------------
extern "C" void kernel_launch(void* const* d_in, const int* in_sizes, int n_in,
                              void* d_out, int out_size, void* d_ws, size_t ws_size,
                              hipStream_t stream) {
    const float* x = (const float*)d_in[0];
    float* out = (float*)d_out;
    float* rn = (float*)d_ws;   // NBATCH*NSEQ floats = 64 KB of scratch

    const int rows = NBATCH * NSEQ;
    rownorm_kernel<<<rows / 8, 256, 0, stream>>>(x, rn);

    const int nblocks = NBATCH * (NSEQ / BM);   // 8 * 64 = 512 workgroups
    contranorm_kernel<<<nblocks, 256, 0, stream>>>(x, rn, out);
}
